// GATModelTune_76647986364938
// MI455X (gfx1250) — compile-verified, matrix operand order checked by reading, not run on hardware
//
#include <hip/hip_runtime.h>
#include <hip/hip_bf16.h>

// ---------------------------------------------------------------------------
// Problem constants (match reference)
// ---------------------------------------------------------------------------
#define NN   50000
#define EE   800000
#define EP   (EE + NN)      // edges + self loops = 850000
#define GG   512
#define DIM  128
#define NEG_SLOPE 0.2f

typedef __attribute__((ext_vector_type(16))) _Float16 v16h;
typedef __attribute__((ext_vector_type(8)))  float    v8f;

union AFrag { v16h v; unsigned u[8]; };

// ordered-uint transform so float max can use atomicMax(unsigned)
__device__ __forceinline__ unsigned f2ord(float f) {
    unsigned u = __float_as_uint(f);
    return (u & 0x80000000u) ? ~u : (u | 0x80000000u);
}
__device__ __forceinline__ float ord2f(unsigned u) {
    return (u & 0x80000000u) ? __uint_as_float(u & 0x7fffffffu)
                             : __uint_as_float(~u);
}

__device__ __forceinline__ float waveReduce(float p) {
    #pragma unroll
    for (int o = 16; o > 0; o >>= 1) p += __shfl_down(p, o, 32);
    return p;
}

// ---------------------------------------------------------------------------
// Encoders
// ---------------------------------------------------------------------------
__global__ void __launch_bounds__(256)
atom_encoder(const int* __restrict__ x, const float* __restrict__ emb,
             _Float16* __restrict__ h16)
{
    int n    = (blockIdx.x * 256 + threadIdx.x) >> 5;
    int lane = threadIdx.x & 31;
    if (n >= NN) return;
    #pragma unroll
    for (int i = 0; i < 4; ++i) {
        int c = i * 32 + lane;
        float acc = 0.f;
        #pragma unroll
        for (int f = 0; f < 9; ++f) {
            int idx = x[n * 9 + f];
            acc += emb[((size_t)(f * 128 + idx)) * DIM + c];
        }
        h16[(size_t)n * DIM + c] = (_Float16)acc;
    }
}

__global__ void __launch_bounds__(256)
bond_encoder(const int* __restrict__ ei, const int* __restrict__ ea,
             const float* __restrict__ emb, _Float16* __restrict__ ea16,
             float* __restrict__ loopsum, float* __restrict__ loopcnt)
{
    int e    = (blockIdx.x * 256 + threadIdx.x) >> 5;
    int lane = threadIdx.x & 31;
    if (e >= EE) return;
    int dst = ei[EE + e];
    #pragma unroll
    for (int i = 0; i < 4; ++i) {
        int c = i * 32 + lane;
        float acc = 0.f;
        #pragma unroll
        for (int f = 0; f < 3; ++f) {
            int idx = ea[e * 3 + f];
            acc += emb[((size_t)(f * 8 + idx)) * DIM + c];
        }
        ea16[(size_t)e * DIM + c] = (_Float16)acc;
        atomicAdd(&loopsum[(size_t)dst * DIM + c], acc);
    }
    if (lane == 0) atomicAdd(&loopcnt[dst], 1.0f);
}

__global__ void __launch_bounds__(256)
self_loop_rows(const float* __restrict__ loopsum, const float* __restrict__ loopcnt,
               _Float16* __restrict__ ea16)
{
    int n    = (blockIdx.x * 256 + threadIdx.x) >> 5;
    int lane = threadIdx.x & 31;
    if (n >= NN) return;
    float inv = 1.0f / fmaxf(loopcnt[n], 1.0f);
    #pragma unroll
    for (int i = 0; i < 4; ++i) {
        int c = i * 32 + lane;
        ea16[(size_t)(EE + n) * DIM + c] =
            (_Float16)(loopsum[(size_t)n * DIM + c] * inv);
    }
}

// ---------------------------------------------------------------------------
// Y[M x 128] = X16[M x 128] @ W[128 x 128] (+bias)   via v_wmma_f32_16x16x32_f16
// 256 threads = 8 waves, each wave owns 16 rows x all 128 cols.
// W is transposed into LDS as f16 (pitch 132 halfwords to break bank conflicts).
// ---------------------------------------------------------------------------
#define WP 132
__global__ void __launch_bounds__(256)
gemm128(const _Float16* __restrict__ X, const float* __restrict__ W,
        const float* __restrict__ bias, int M,
        float* __restrict__ Yf, _Float16* __restrict__ Yh)
{
    __shared__ _Float16 Wt[128 * WP];
    const int t = threadIdx.x;
    for (int i = t; i < 128 * 128; i += 256) {
        int k = i >> 7, o = i & 127;          // coalesced read of W[k][o]
        Wt[o * WP + k] = (_Float16)W[i];      // transposed store
    }
    __syncthreads();

    const int lane = t & 31;
    const int wave = t >> 5;
    const int half = lane >> 4;
    const int l16  = lane & 15;
    const int rowBase = blockIdx.x * 128 + wave * 16;
    const int row = rowBase + l16;
    const bool rowOK = row < M;

    v8f acc[8] = {};

    for (int kc = 0; kc < 128; kc += 32) {
        AFrag a;
        if (rowOK) {
            const unsigned* xp = (const unsigned*)(X + (size_t)row * DIM + kc);
            #pragma unroll
            for (int j = 0; j < 8; ++j) {
                int k0 = (j < 4) ? (half * 8 + 2 * j) : (16 + half * 8 + 2 * (j - 4));
                a.u[j] = xp[k0 >> 1];
            }
        } else {
            #pragma unroll
            for (int j = 0; j < 8; ++j) a.u[j] = 0u;
        }
        #pragma unroll
        for (int nt = 0; nt < 8; ++nt) {
            AFrag b;
            const unsigned* wp = (const unsigned*)(Wt + (nt * 16 + l16) * WP + kc);
            #pragma unroll
            for (int j = 0; j < 8; ++j) {
                int k0 = (j < 4) ? (half * 8 + 2 * j) : (16 + half * 8 + 2 * (j - 4));
                b.u[j] = wp[k0 >> 1];
            }
            acc[nt] = __builtin_amdgcn_wmma_f32_16x16x32_f16(
                false, a.v, false, b.v, (short)0, acc[nt], false, false);
        }
    }

    #pragma unroll
    for (int nt = 0; nt < 8; ++nt) {
        int col = nt * 16 + l16;
        float bv = bias ? bias[col] : 0.0f;
        #pragma unroll
        for (int r = 0; r < 8; ++r) {
            int rr = rowBase + half * 8 + r;
            if (rr < M) {
                float v = acc[nt][r] + bv;
                if (Yf) Yf[(size_t)rr * DIM + col] = v;
                else    Yh[(size_t)rr * DIM + col] = (_Float16)v;
            }
        }
    }
}

// ---------------------------------------------------------------------------
// GATv2 edge passes
// ---------------------------------------------------------------------------
__global__ void __launch_bounds__(256)
edge_scores(const int* __restrict__ ei,
            const float* __restrict__ hl, const float* __restrict__ hr,
            const _Float16* __restrict__ eW, const float* __restrict__ att,
            float* __restrict__ s, unsigned* __restrict__ m)
{
    int e    = (blockIdx.x * 256 + threadIdx.x) >> 5;
    int lane = threadIdx.x & 31;
    if (e >= EP) return;
    int src, dst;
    if (e < EE) { src = ei[e]; dst = ei[EE + e]; } else { src = dst = e - EE; }
    float p = 0.f;
    #pragma unroll
    for (int i = 0; i < 4; ++i) {
        int c = i * 32 + lane;
        float z = hl[(size_t)src * DIM + c] + hr[(size_t)dst * DIM + c]
                + (float)eW[(size_t)e * DIM + c];
        z = (z > 0.f) ? z : NEG_SLOPE * z;       // leaky_relu
        p += z * att[c];
    }
    p = waveReduce(p);
    if (lane == 0) {
        s[e] = p;
        atomicMax(&m[dst], f2ord(p));
    }
}

__global__ void __launch_bounds__(256)
edge_expsum(const int* __restrict__ ei, float* __restrict__ s,
            const unsigned* __restrict__ m, float* __restrict__ denom)
{
    int e = blockIdx.x * 256 + threadIdx.x;
    if (e >= EP) return;
    int dst = (e < EE) ? ei[EE + e] : (e - EE);
    float ex = __expf(s[e] - ord2f(m[dst]));
    s[e] = ex;
    atomicAdd(&denom[dst], ex);
}

__global__ void __launch_bounds__(256)
edge_scatter(const int* __restrict__ ei, const float* __restrict__ s,
             const float* __restrict__ denom, const float* __restrict__ hl,
             float* __restrict__ out)
{
    int e    = (blockIdx.x * 256 + threadIdx.x) >> 5;
    int lane = threadIdx.x & 31;
    if (e >= EP) return;
    int src, dst;
    if (e < EE) { src = ei[e]; dst = ei[EE + e]; } else { src = dst = e - EE; }
    float alpha = s[e] / denom[dst];
    #pragma unroll
    for (int i = 0; i < 4; ++i) {
        int c = i * 32 + lane;
        atomicAdd(&out[(size_t)dst * DIM + c],
                  alpha * hl[(size_t)src * DIM + c]);
    }
}

// h16 = f16(relu(out + bias))   (layer-1 finalize, feeds layer-2 GEMMs)
__global__ void __launch_bounds__(256)
finalize_relu(const float* __restrict__ out, const float* __restrict__ bias,
              _Float16* __restrict__ h16)
{
    size_t idx = (size_t)blockIdx.x * 256 + threadIdx.x;
    if (idx >= (size_t)NN * DIM) return;
    int c = (int)(idx & (DIM - 1));
    h16[idx] = (_Float16)fmaxf(out[idx] + bias[c], 0.0f);
}

// layer-2 finalize fused with mean-pool accumulation
__global__ void __launch_bounds__(256)
pool_accum(const float* __restrict__ out, const float* __restrict__ bias,
           const int* __restrict__ batch, float* __restrict__ pool,
           float* __restrict__ cntg)
{
    int n    = (blockIdx.x * 256 + threadIdx.x) >> 5;
    int lane = threadIdx.x & 31;
    if (n >= NN) return;
    int g = batch[n];
    #pragma unroll
    for (int i = 0; i < 4; ++i) {
        int c = i * 32 + lane;
        atomicAdd(&pool[(size_t)g * DIM + c], out[(size_t)n * DIM + c] + bias[c]);
    }
    if (lane == 0) atomicAdd(&cntg[g], 1.0f);
}

__global__ void __launch_bounds__(256)
readout(const float* __restrict__ pool, const float* __restrict__ cntg,
        const float* __restrict__ Wout, const float* __restrict__ bout,
        float* __restrict__ y)
{
    int g    = (blockIdx.x * 256 + threadIdx.x) >> 5;
    int lane = threadIdx.x & 31;
    if (g >= GG) return;
    float inv = 1.0f / fmaxf(cntg[g], 1.0f);
    float p = 0.f;
    #pragma unroll
    for (int i = 0; i < 4; ++i) {
        int c = i * 32 + lane;
        p += pool[(size_t)g * DIM + c] * inv * Wout[c];
    }
    p = waveReduce(p);
    if (lane == 0) y[g] = 1.0f / (1.0f + __expf(-(p + bout[0])));
}

// ---------------------------------------------------------------------------
// Host-side launch
// ---------------------------------------------------------------------------
static inline void* ws_take(char*& p, size_t bytes) {
    void* r = (void*)p;
    p += (bytes + 255) & ~(size_t)255;
    return r;
}
static inline unsigned cdiv(size_t a, size_t b) { return (unsigned)((a + b - 1) / b); }

extern "C" void kernel_launch(void* const* d_in, const int* in_sizes, int n_in,
                              void* d_out, int out_size, void* d_ws, size_t ws_size,
                              hipStream_t stream) {
    (void)in_sizes; (void)n_in; (void)out_size; (void)ws_size;
    const int*   x        = (const int*)  d_in[0];
    const int*   ei       = (const int*)  d_in[1];
    const int*   eattr    = (const int*)  d_in[2];
    const int*   batch    = (const int*)  d_in[3];
    const float* atom_emb = (const float*)d_in[4];
    const float* bond_emb = (const float*)d_in[5];
    const float* W1l = (const float*)d_in[6];  const float* b1l  = (const float*)d_in[7];
    const float* W1r = (const float*)d_in[8];  const float* b1r  = (const float*)d_in[9];
    const float* W1e = (const float*)d_in[10]; const float* att1 = (const float*)d_in[11];
    const float* bias1 = (const float*)d_in[12];
    const float* W2l = (const float*)d_in[13]; const float* b2l  = (const float*)d_in[14];
    const float* W2r = (const float*)d_in[15]; const float* b2r  = (const float*)d_in[16];
    const float* W2e = (const float*)d_in[17]; const float* att2 = (const float*)d_in[18];
    const float* bias2 = (const float*)d_in[19];
    const float* Wout = (const float*)d_in[20]; const float* bout = (const float*)d_in[21];
    float* y = (float*)d_out;

    char* p = (char*)d_ws;
    _Float16* h16     = (_Float16*)ws_take(p, (size_t)NN * DIM * 2);
    _Float16* ea16    = (_Float16*)ws_take(p, (size_t)EP * DIM * 2);
    _Float16* eW16    = (_Float16*)ws_take(p, (size_t)EP * DIM * 2);
    float*    hl      = (float*)   ws_take(p, (size_t)NN * DIM * 4);
    float*    hr      = (float*)   ws_take(p, (size_t)NN * DIM * 4);
    float*    out     = (float*)   ws_take(p, (size_t)NN * DIM * 4);
    float*    loopsum = (float*)   ws_take(p, (size_t)NN * DIM * 4);
    float*    loopcnt = (float*)   ws_take(p, (size_t)NN * 4);
    float*    s       = (float*)   ws_take(p, (size_t)EP * 4);
    unsigned* m       = (unsigned*)ws_take(p, (size_t)NN * 4);
    float*    denom   = (float*)   ws_take(p, (size_t)NN * 4);
    float*    pool    = (float*)   ws_take(p, (size_t)GG * DIM * 4);
    float*    cntg    = (float*)   ws_take(p, (size_t)GG * 4);

    const unsigned gbN  = cdiv(NN, 8);               // wave-per-node kernels
    const unsigned gbE  = cdiv(EE, 8);               // wave-per-edge (E)
    const unsigned gbEP = cdiv(EP, 8);               // wave-per-edge (E')
    const unsigned gbEPt = cdiv(EP, 256);            // thread-per-edge

    // ---- encoders -----------------------------------------------------
    hipMemsetAsync(loopsum, 0, (size_t)NN * DIM * 4, stream);
    hipMemsetAsync(loopcnt, 0, (size_t)NN * 4, stream);
    atom_encoder<<<gbN, 256, 0, stream>>>(x, atom_emb, h16);
    bond_encoder<<<gbE, 256, 0, stream>>>(ei, eattr, bond_emb, ea16, loopsum, loopcnt);
    self_loop_rows<<<gbN, 256, 0, stream>>>(loopsum, loopcnt, ea16);

    const float* Wl[2]  = {W1l, W2l};  const float* bl[2]  = {b1l, b2l};
    const float* Wr[2]  = {W1r, W2r};  const float* br[2]  = {b1r, b2r};
    const float* We[2]  = {W1e, W2e};  const float* att[2] = {att1, att2};
    const float* bia[2] = {bias1, bias2};

    for (int L = 0; L < 2; ++L) {
        // dense transforms (WMMA)
        gemm128<<<cdiv(NN, 128), 256, 0, stream>>>(h16, Wl[L], bl[L], NN, hl, nullptr);
        gemm128<<<cdiv(NN, 128), 256, 0, stream>>>(h16, Wr[L], br[L], NN, hr, nullptr);
        gemm128<<<cdiv(EP, 128), 256, 0, stream>>>(ea16, We[L], nullptr, EP, nullptr, eW16);

        // segment softmax + aggregate
        hipMemsetAsync(m,     0, (size_t)NN * 4, stream);
        hipMemsetAsync(denom, 0, (size_t)NN * 4, stream);
        hipMemsetAsync(out,   0, (size_t)NN * DIM * 4, stream);
        edge_scores <<<gbEP,  256, 0, stream>>>(ei, hl, hr, eW16, att[L], s, m);
        edge_expsum <<<gbEPt, 256, 0, stream>>>(ei, s, m, denom);
        edge_scatter<<<gbEP,  256, 0, stream>>>(ei, s, denom, hl, out);

        if (L == 0) {
            finalize_relu<<<cdiv((size_t)NN * DIM, 256), 256, 0, stream>>>(out, bia[0], h16);
        } else {
            hipMemsetAsync(pool, 0, (size_t)GG * DIM * 4, stream);
            hipMemsetAsync(cntg, 0, (size_t)GG * 4, stream);
            pool_accum<<<gbN, 256, 0, stream>>>(out, bia[1], batch, pool, cntg);
            readout<<<cdiv(GG, 8), 256, 0, stream>>>(pool, cntg, Wout, bout, y);
        }
    }
}
